// MoCoGraph_55654186222289
// MI455X (gfx1250) — matrix operand order repaired and checked by reading, not compiled, not run on hardware
//
#include <hip/hip_runtime.h>
#include <hip/hip_bf16.h>

// ---------------- problem constants (match setup_inputs) ----------------
constexpr int N_  = 10000;     // nodes
constexpr int G_  = 1000;      // input feats
constexpr int GP_ = 1024;      // padded K for layer-1 GEMM
constexpr int E_  = 160000;    // edges (before self loops)
constexpr int E2_ = E_ + N_;   // edges incl self loops
constexpr int H_  = 4;         // heads
constexpr int C_  = 128;       // out channels
constexpr int HC_ = H_ * C_;   // 512
constexpr int B_  = 1024;      // seed nodes
constexpr int QK_ = 1024;      // queue columns
constexpr float NEG_SLOPE = 0.2f;
constexpr float INV_TEMP  = 5.0f;   // 1/0.2
constexpr float MOM       = 0.99f;

typedef __attribute__((ext_vector_type(16))) _Float16 v16h;
typedef __attribute__((ext_vector_type(8)))  _Float16 v8h;   // 16-byte chunk
typedef __attribute__((ext_vector_type(8)))  float    v8f;

// ---------------- small helpers ----------------
__device__ __forceinline__ unsigned f2o(float f) {           // order-preserving f32->u32
  unsigned u = __float_as_uint(f);
  return (u & 0x80000000u) ? ~u : (u | 0x80000000u);
}
__device__ __forceinline__ float o2f(unsigned u) {
  return (u & 0x80000000u) ? __uint_as_float(u & 0x7FFFFFFFu) : __uint_as_float(~u);
}

// ---------------- generic elementwise kernels ----------------
__global__ void k_lerp(float* dst, const float* pk, const float* pq, int n) {
  int i = blockIdx.x * blockDim.x + threadIdx.x;
  if (i < n) dst[i] = MOM * pk[i] + (1.0f - MOM) * pq[i];
}
__global__ void k_cvt_f16(_Float16* dst, const float* src, int n) {
  int i = blockIdx.x * blockDim.x + threadIdx.x;
  if (i < n) dst[i] = (_Float16)src[i];
}
// x [N x G] f32 -> [N x GP] f16, zero-padded columns
__global__ void k_cvt_pad_cols(_Float16* dst, const float* src) {
  int i = blockIdx.x * blockDim.x + threadIdx.x;
  if (i >= N_ * GP_) return;
  int r = i >> 10, c = i & (GP_ - 1);
  dst[i] = (c < G_) ? (_Float16)src[r * G_ + c] : (_Float16)0.0f;
}
// W1 [G x HC] f32 -> [GP x HC] f16, zero-padded rows
__global__ void k_cvt_pad_rows(_Float16* dst, const float* src) {
  int i = blockIdx.x * blockDim.x + threadIdx.x;
  if (i >= GP_ * HC_) return;
  int r = i / HC_;
  dst[i] = (r < G_) ? (_Float16)src[i] : (_Float16)0.0f;
}
__global__ void k_fill_f32(float* p, float v, int n) {
  int i = blockIdx.x * blockDim.x + threadIdx.x;
  if (i < n) p[i] = v;
}
__global__ void k_init_seg(unsigned* emax, float* den, int n) {
  int i = blockIdx.x * blockDim.x + threadIdx.x;
  if (i < n) { emax[i] = f2o(-3.402823466e38f); den[i] = 0.0f; }
}

// ---------------- WMMA GEMM: C[ldc] = alpha * (A f16 [MxK]) x (B f16 [KxNc]) ----------------
// 256 threads = 8 waves. Each wave computes a 16x64 strip: 4 accumulators share one
// A fragment per k-step (A-reuse quadruples WMMA work per A load).
// Requires: K % 32 == 0, Nc % 512 == 0, M % 16 == 0.
__global__ void k_gemm_wmma(const _Float16* __restrict__ A,
                            const _Float16* __restrict__ Bm,
                            float* __restrict__ Cm,
                            int M, int K, int Nc, int ldc, float alpha) {
  const int wave = threadIdx.x >> 5;
  const int lane = threadIdx.x & 31;
  const int tileN = blockIdx.x * 512 + wave * 64;
  const int tileM = blockIdx.y * 16;
  if (tileN >= Nc || tileM >= M) return;     // wave-uniform: EXEC stays all-ones
  const int r  = lane & 15;
  const int hi = lane >> 4;

  v8f acc[4] = {{}, {}, {}, {}};
  const _Float16* arow0 = A  + (size_t)(tileM + r) * K + hi * 8;
  const _Float16* brow0 = Bm + tileN + r;

  for (int k0 = 0; k0 < K; k0 += 32) {
    const _Float16* arow = arow0 + k0;
    const _Float16* brow = brow0 + (size_t)k0 * Nc;
    // A fragment: halves 0..7 -> K = k0+hi*8+{0..7}; halves 8..15 -> +16
    v8h a0 = *(const v8h*)(arow);
    v8h a1 = *(const v8h*)(arow + 16);
    if (k0 + 32 < K) {                        // stage next K-tile into the WGP caches
      __builtin_prefetch(arow + 32, 0, 3);
      __builtin_prefetch(brow + (size_t)32 * Nc, 0, 3);
    }
    v16h af;
#pragma unroll
    for (int i = 0; i < 8; ++i) { af[i] = a0[i]; af[i + 8] = a1[i]; }
    // 4 B tiles, 4 WMMAs, one shared A fragment
#pragma unroll
    for (int t = 0; t < 4; ++t) {
      const _Float16* bp = brow + t * 16;     // lane holds B col r, K = k0+hi*8+{0..7}, +16
      v16h bf;
#pragma unroll
      for (int i = 0; i < 8; ++i) { bf[i] = bp[(size_t)(hi * 8 + i) * Nc]; bf[i + 8] = bp[(size_t)(16 + hi * 8 + i) * Nc]; }
      acc[t] = __builtin_amdgcn_wmma_f32_16x16x32_f16(false, af, false, bf,
                                                      (short)0, acc[t], false, false);
    }
  }
  // D layout: lanes 0-15 -> N=lane, M=v; lanes 16-31 -> N=lane-16, M=v+8
#pragma unroll
  for (int t = 0; t < 4; ++t)
#pragma unroll
    for (int v = 0; v < 8; ++v)
      Cm[(size_t)(tileM + hi * 8 + v) * ldc + tileN + t * 16 + r] = alpha * acc[t][v];
}

// ---------------- attention coefficients: a_src/a_dst [N,H] ----------------
__global__ void k_att(const float* __restrict__ Hb, const float* __restrict__ as,
                      const float* __restrict__ ad, float* asrc, float* adst) {
  int gw   = (blockIdx.x * blockDim.x + threadIdx.x) >> 5;
  int lane = threadIdx.x & 31;
  if (gw >= N_ * H_) return;
  int n = gw >> 2, h = gw & 3;
  const float4* hv = (const float4*)(Hb + (size_t)n * HC_ + h * C_);
  const float4* av = (const float4*)(as + h * C_);
  const float4* dv = (const float4*)(ad + h * C_);
  float4 x = hv[lane], a = av[lane], d = dv[lane];
  float ss = x.x * a.x + x.y * a.y + x.z * a.z + x.w * a.w;
  float sd = x.x * d.x + x.y * d.y + x.z * d.z + x.w * d.w;
  for (int o = 16; o > 0; o >>= 1) { ss += __shfl_down(ss, o, 32); sd += __shfl_down(sd, o, 32); }
  if (lane == 0) { asrc[gw] = ss; adst[gw] = sd; }
}

// ---------------- edge passes ----------------
__global__ void k_edge_logits(const int* __restrict__ ei, const float* asrc,
                              const float* adst, float* ebuf, unsigned* emax) {
  int idx = blockIdx.x * blockDim.x + threadIdx.x;
  if (idx >= E2_ * H_) return;
  int e = idx >> 2, h = idx & 3;
  int s, d;
  if (e < E_) { s = ei[e]; d = ei[E_ + e]; } else { s = d = e - E_; }
  float x = asrc[s * H_ + h] + adst[d * H_ + h];
  float ev = x > 0.0f ? x : NEG_SLOPE * x;
  ebuf[idx] = ev;
  atomicMax(&emax[d * H_ + h], f2o(ev));
}
__global__ void k_edge_exp(const int* __restrict__ ei, const unsigned* emax,
                           float* ebuf, float* den) {
  int idx = blockIdx.x * blockDim.x + threadIdx.x;
  if (idx >= E2_ * H_) return;
  int e = idx >> 2, h = idx & 3;
  int d = (e < E_) ? ei[E_ + e] : (e - E_);
  float m  = o2f(emax[d * H_ + h]);
  float ee = expf(ebuf[idx] - m);
  ebuf[idx] = ee;
  atomicAdd(&den[d * H_ + h], ee);
}
__global__ void k_edge_msg(const int* __restrict__ ei, const float* __restrict__ Hb,
                           const float* __restrict__ ebuf, const float* __restrict__ den,
                           float* __restrict__ Agg) {
  int idx = blockIdx.x * blockDim.x + threadIdx.x;
  if (idx >= E2_ * HC_) return;
  int e = idx / HC_, rem = idx % HC_, h = rem >> 7;
  int s, d;
  if (e < E_) { s = ei[e]; d = ei[E_ + e]; } else { s = d = e - E_; }
  float alpha = ebuf[e * H_ + h] / (den[d * H_ + h] + 1e-16f);
  atomicAdd(&Agg[(size_t)d * HC_ + rem], Hb[(size_t)s * HC_ + rem] * alpha);
}

// ---------------- post ops ----------------
__global__ void k_post_concat_relu(const float* Agg, const float* b, _Float16* out16) {
  int i = blockIdx.x * blockDim.x + threadIdx.x;
  if (i >= N_ * HC_) return;
  float v = Agg[i] + b[i & (HC_ - 1)];
  out16[i] = (_Float16)(v > 0.0f ? v : 0.0f);
}
__global__ void k_post_mean(const float* Agg, const float* b, float* out) {
  int i = blockIdx.x * blockDim.x + threadIdx.x;
  if (i >= N_ * C_) return;
  int n = i >> 7, c = i & (C_ - 1);
  const float* row = Agg + (size_t)n * HC_;
  out[i] = 0.25f * (row[c] + row[C_ + c] + row[2 * C_ + c] + row[3 * C_ + c]) + b[c];
}
__global__ void k_l2norm(const float* in, float* out) {
  int row  = (blockIdx.x * blockDim.x + threadIdx.x) >> 5;
  int lane = threadIdx.x & 31;
  if (row >= B_) return;
  float4 v = ((const float4*)(in + (size_t)row * C_))[lane];
  float s = v.x * v.x + v.y * v.y + v.z * v.z + v.w * v.w;
  for (int o = 16; o > 0; o >>= 1) s += __shfl_down(s, o, 32);
  s = __shfl(s, 0, 32);
  float inv = rsqrtf(s + 1e-24f);
  float4 w = {v.x * inv, v.y * inv, v.z * inv, v.w * inv};
  ((float4*)(out + (size_t)row * C_))[lane] = w;
}
__global__ void k_lpos(const float* q, const float* k, float* out) {
  int row  = (blockIdx.x * blockDim.x + threadIdx.x) >> 5;
  int lane = threadIdx.x & 31;
  if (row >= B_) return;
  float4 a = ((const float4*)(q + (size_t)row * C_))[lane];
  float4 b = ((const float4*)(k + (size_t)row * C_))[lane];
  float s = a.x * b.x + a.y * b.y + a.z * b.z + a.w * b.w;
  for (int o = 16; o > 0; o >>= 1) s += __shfl_down(s, o, 32);
  if (lane == 0) out[(size_t)row * (QK_ + 1)] = s * INV_TEMP;
}
__global__ void k_labels(float* out) {
  int i = blockIdx.x * blockDim.x + threadIdx.x;
  if (i < B_) out[(size_t)B_ * (QK_ + 1) + i] = 0.0f;
}

// ---------------- host side ----------------
struct WsPtrs {
  _Float16* A16; _Float16* B16;
  float* Hb; float* Agg; float* asrc; float* adst;
  unsigned* emax; float* den; float* ebuf; float* out2;
};
static inline int cdiv(long long a, int b) { return (int)((a + b - 1) / b); }

static void run_encoder(const float* x,
                        const float* W1, const float* as1, const float* ad1, const float* b1,
                        const float* W2, const float* as2, const float* ad2, const float* b2,
                        const int* ei, const WsPtrs& w, float* seed, hipStream_t s) {
  const int T = 256;
  // ---- layer 1 ----
  k_cvt_pad_cols<<<cdiv((long long)N_ * GP_, T), T, 0, s>>>(w.A16, x);
  k_cvt_pad_rows<<<cdiv((long long)GP_ * HC_, T), T, 0, s>>>(w.B16, W1);
  k_gemm_wmma<<<dim3(HC_ / 512, N_ / 16), T, 0, s>>>(w.A16, w.B16, w.Hb, N_, GP_, HC_, HC_, 1.0f);
  k_att<<<cdiv((long long)N_ * H_ * 32, T), T, 0, s>>>(w.Hb, as1, ad1, w.asrc, w.adst);
  k_init_seg<<<cdiv(N_ * H_, T), T, 0, s>>>(w.emax, w.den, N_ * H_);
  k_fill_f32<<<cdiv((long long)N_ * HC_, T), T, 0, s>>>(w.Agg, 0.0f, N_ * HC_);
  k_edge_logits<<<cdiv((long long)E2_ * H_, T), T, 0, s>>>(ei, w.asrc, w.adst, w.ebuf, w.emax);
  k_edge_exp<<<cdiv((long long)E2_ * H_, T), T, 0, s>>>(ei, w.emax, w.ebuf, w.den);
  k_edge_msg<<<cdiv((long long)E2_ * HC_, T), T, 0, s>>>(ei, w.Hb, w.ebuf, w.den, w.Agg);
  k_post_concat_relu<<<cdiv((long long)N_ * HC_, T), T, 0, s>>>(w.Agg, b1, w.A16); // f16 h for layer 2
  // ---- layer 2 ----
  k_cvt_f16<<<cdiv((long long)HC_ * HC_, T), T, 0, s>>>(w.B16, W2, HC_ * HC_);
  k_gemm_wmma<<<dim3(HC_ / 512, N_ / 16), T, 0, s>>>(w.A16, w.B16, w.Hb, N_, HC_, HC_, HC_, 1.0f);
  k_att<<<cdiv((long long)N_ * H_ * 32, T), T, 0, s>>>(w.Hb, as2, ad2, w.asrc, w.adst);
  k_init_seg<<<cdiv(N_ * H_, T), T, 0, s>>>(w.emax, w.den, N_ * H_);
  k_fill_f32<<<cdiv((long long)N_ * HC_, T), T, 0, s>>>(w.Agg, 0.0f, N_ * HC_);
  k_edge_logits<<<cdiv((long long)E2_ * H_, T), T, 0, s>>>(ei, w.asrc, w.adst, w.ebuf, w.emax);
  k_edge_exp<<<cdiv((long long)E2_ * H_, T), T, 0, s>>>(ei, w.emax, w.ebuf, w.den);
  k_edge_msg<<<cdiv((long long)E2_ * HC_, T), T, 0, s>>>(ei, w.Hb, w.ebuf, w.den, w.Agg);
  k_post_mean<<<cdiv((long long)N_ * C_, T), T, 0, s>>>(w.Agg, b2, w.out2);
  k_l2norm<<<cdiv((long long)B_ * 32, T), T, 0, s>>>(w.out2, seed);
}

extern "C" void kernel_launch(void* const* d_in, const int* in_sizes, int n_in,
                              void* d_out, int out_size, void* d_ws, size_t ws_size,
                              hipStream_t stream) {
  const float* im_q = (const float*)d_in[0];
  const float* im_k = (const float*)d_in[1];
  const int*   ei   = (const int*)d_in[2];
  const float* Wq1 = (const float*)d_in[4],  *asq1 = (const float*)d_in[5];
  const float* adq1 = (const float*)d_in[6], *bq1 = (const float*)d_in[7];
  const float* Wq2 = (const float*)d_in[8],  *asq2 = (const float*)d_in[9];
  const float* adq2 = (const float*)d_in[10], *bq2 = (const float*)d_in[11];
  const float* Wk1 = (const float*)d_in[12], *ask1 = (const float*)d_in[13];
  const float* adk1 = (const float*)d_in[14], *bk1 = (const float*)d_in[15];
  const float* Wk2 = (const float*)d_in[16], *ask2 = (const float*)d_in[17];
  const float* adk2 = (const float*)d_in[18], *bk2 = (const float*)d_in[19];
  const float* queue = (const float*)d_in[20];
  float* outf = (float*)d_out;

  char* ws = (char*)d_ws;
  size_t off = 0;
  auto alloc = [&](size_t bytes) -> void* {
    void* p = ws + off; off += (bytes + 255) & ~(size_t)255; return p;
  };
  // momentum-updated key params
  float* kpW1 = (float*)alloc((size_t)G_ * HC_ * 4);
  float* kpa1 = (float*)alloc(HC_ * 4);
  float* kpd1 = (float*)alloc(HC_ * 4);
  float* kpb1 = (float*)alloc(HC_ * 4);
  float* kpW2 = (float*)alloc((size_t)HC_ * HC_ * 4);
  float* kpa2 = (float*)alloc(HC_ * 4);
  float* kpd2 = (float*)alloc(HC_ * 4);
  float* kpb2 = (float*)alloc(C_ * 4);
  // encoder scratch
  WsPtrs w;
  w.A16  = (_Float16*)alloc((size_t)N_ * GP_ * 2);
  w.B16  = (_Float16*)alloc((size_t)GP_ * HC_ * 2);
  w.Hb   = (float*)alloc((size_t)N_ * HC_ * 4);
  w.Agg  = (float*)alloc((size_t)N_ * HC_ * 4);
  w.asrc = (float*)alloc((size_t)N_ * H_ * 4);
  w.adst = (float*)alloc((size_t)N_ * H_ * 4);
  w.emax = (unsigned*)alloc((size_t)N_ * H_ * 4);
  w.den  = (float*)alloc((size_t)N_ * H_ * 4);
  w.ebuf = (float*)alloc((size_t)E2_ * H_ * 4);
  w.out2 = (float*)alloc((size_t)N_ * C_ * 4);
  float* qseed = (float*)alloc((size_t)B_ * C_ * 4);
  float* kseed = (float*)alloc((size_t)B_ * C_ * 4);
  _Float16* q16 = (_Float16*)alloc((size_t)B_ * C_ * 2);
  _Float16* qq16 = (_Float16*)alloc((size_t)C_ * QK_ * 2);

  const int T = 256;
  // query encoder
  run_encoder(im_q, Wq1, asq1, adq1, bq1, Wq2, asq2, adq2, bq2, ei, w, qseed, stream);
  // momentum update of key params
  k_lerp<<<cdiv((long long)G_ * HC_, T), T, 0, stream>>>(kpW1, Wk1, Wq1, G_ * HC_);
  k_lerp<<<cdiv(HC_, T), T, 0, stream>>>(kpa1, ask1, asq1, HC_);
  k_lerp<<<cdiv(HC_, T), T, 0, stream>>>(kpd1, adk1, adq1, HC_);
  k_lerp<<<cdiv(HC_, T), T, 0, stream>>>(kpb1, bk1, bq1, HC_);
  k_lerp<<<cdiv((long long)HC_ * HC_, T), T, 0, stream>>>(kpW2, Wk2, Wq2, HC_ * HC_);
  k_lerp<<<cdiv(HC_, T), T, 0, stream>>>(kpa2, ask2, asq2, HC_);
  k_lerp<<<cdiv(HC_, T), T, 0, stream>>>(kpd2, adk2, adq2, HC_);
  k_lerp<<<cdiv(C_, T), T, 0, stream>>>(kpb2, bk2, bq2, C_);
  // key encoder
  run_encoder(im_k, kpW1, kpa1, kpd1, kpb1, kpW2, kpa2, kpd2, kpb2, ei, w, kseed, stream);
  // logits: l_neg = (qseed @ queue) / TEMP written into out[:,1:1025]
  k_cvt_f16<<<cdiv((long long)B_ * C_, T), T, 0, stream>>>(q16, qseed, B_ * C_);
  k_cvt_f16<<<cdiv((long long)C_ * QK_, T), T, 0, stream>>>(qq16, queue, C_ * QK_);
  k_gemm_wmma<<<dim3(QK_ / 512, B_ / 16), T, 0, stream>>>(q16, qq16, outf + 1,
                                                          B_, C_, QK_, QK_ + 1, INV_TEMP);
  // l_pos in column 0, labels appended as zeros
  k_lpos<<<cdiv((long long)B_ * 32, T), T, 0, stream>>>(qseed, kseed, outf);
  k_labels<<<cdiv(B_, T), T, 0, stream>>>(outf);
}
